// MessageGraphBuilder_5549097746993
// MI455X (gfx1250) — compile-verified, matrix operand order checked by reading, not run on hardware
//
#include <hip/hip_runtime.h>
#include <hip/hip_bf16.h>

// ---------------------------------------------------------------------------
// MessageGraphBuilder for MI455X (gfx1250, wave32, WMMA)
//   N = 8192 messages, D = 256 dims, K_SEM = 5, THRESH = 0.2
//   Fused bf16-WMMA GEMM + streaming masked top-5 (sim matrix never hits HBM)
//   Round 2: register double-buffered B tiles, b128 LDS scan, prefetch
// ---------------------------------------------------------------------------

#define KSEM 5
#define THRESH 0.2f
#define NEG_INF (-1e30f)
#define DIM 256

typedef __bf16 bf16_t;
typedef bf16_t v16bf __attribute__((ext_vector_type(16)));
typedef float  v8f   __attribute__((ext_vector_type(8)));

__device__ __forceinline__ unsigned short f2bf_rne(float f) {
    unsigned u = __float_as_uint(f);
    unsigned r = u + 0x7FFFu + ((u >> 16) & 1u);
    return (unsigned short)(r >> 16);
}

// ---------------- Kernel 1: L2-normalize rows, convert to bf16 -------------
__global__ __launch_bounds__(256) void mg_normalize_bf16(
        const float* __restrict__ emb, unsigned short* __restrict__ out, int N) {
    int row  = blockIdx.x * (blockDim.x >> 5) + (threadIdx.x >> 5);
    int lane = threadIdx.x & 31;
    if (row >= N) return;
    const float* r = emb + (size_t)row * DIM;
    float v[8];
    float ss = 0.f;
#pragma unroll
    for (int t = 0; t < 8; ++t) {
        v[t] = r[lane + 32 * t];
        ss += v[t] * v[t];
    }
#pragma unroll
    for (int off = 16; off > 0; off >>= 1) ss += __shfl_xor(ss, off, 32);
    float inv = 1.0f / fmaxf(sqrtf(ss), 1e-12f);
    unsigned short* o = out + (size_t)row * DIM;
#pragma unroll
    for (int t = 0; t < 8; ++t) o[lane + 32 * t] = f2bf_rne(v[t] * inv);
}

// ---------------- Kernel 2: WMMA sim GEMM + streaming top-5 ----------------
// grid = N/16 workgroups; 128 threads = 4 waves per workgroup.
// Workgroup b owns rows [16b, 16b+16); waves stride over 16-col tiles j < i.
__global__ __launch_bounds__(128) void mg_sim_topk(
        const unsigned short* __restrict__ ebf,
        float* __restrict__ o_ei, float* __restrict__ o_valid,
        int N, int nT, int E) {
    const int i0   = blockIdx.x * 16;
    const int wave = threadIdx.x >> 5;
    const int lane = threadIdx.x & 31;
    const int lrow = lane & 15;   // row (A) / col (B) within tile
    const int half = lane >> 4;   // K-half for fragments; col-half for scan

    __shared__ float ctile[4][16][16];
    __shared__ float svals[16][8][KSEM];
    __shared__ int   sidx[16][8][KSEM];

    // --- A fragments: 16x256 bf16 row band, 8 chunks of K=32, in registers.
    // A k order used here (contiguous, B matches): lanes 0-15 -> M=lane, K=0..15; lanes 16-31 -> K=16..31.
    v16bf afrag[8];
    const unsigned short* abase = ebf + (size_t)(i0 + lrow) * DIM + half * 16;
#pragma unroll
    for (int kc = 0; kc < 8; ++kc)
        afrag[kc] = *(const v16bf*)(abase + kc * 32);

    float tv[KSEM];
    int   ti[KSEM];
#pragma unroll
    for (int k = 0; k < KSEM; ++k) { tv[k] = NEG_INF; ti[k] = -1; }

    // columns needed: j <= i0+13 (j < i and j != i-1)
    const int ntiles = (i0 + 13) / 16 + 1;
    const size_t lane_off = (size_t)lrow * DIM + half * 16;

    int jt = wave;
    v16bf bcur[8];
    if (jt < ntiles) {
        // preload first tile's B fragments (B = E^T: same contiguous pattern
        // as A, keyed by column index)
        const unsigned short* bb = ebf + (size_t)(jt * 16) * DIM + lane_off;
#pragma unroll
        for (int kc = 0; kc < 8; ++kc)
            bcur[kc] = *(const v16bf*)(bb + kc * 32);
    }

    for (; jt < ntiles; jt += 4) {
        const int j0 = jt * 16;

        // ---- issue next tile's loads first (double buffer in registers) ---
        const int jn    = jt + 4;
        const int jload = (jn < ntiles) ? jn : jt;   // clamp: stay in bounds
        const unsigned short* bnb = ebf + (size_t)(jload * 16) * DIM + lane_off;
        v16bf bnext[8];
#pragma unroll
        for (int kc = 0; kc < 8; ++kc)
            bnext[kc] = *(const v16bf*)(bnb + kc * 32);

        // warm L2 two tiles ahead (first sweep comes from HBM)
        if (jt + 8 < ntiles)
            __builtin_prefetch(ebf + (size_t)((jt + 8) * 16) * DIM + lane_off, 0, 1);

        // ---- 8 back-to-back WMMAs on the current tile ---------------------
        v8f acc = {0.f, 0.f, 0.f, 0.f, 0.f, 0.f, 0.f, 0.f};
#pragma unroll
        for (int kc = 0; kc < 8; ++kc)
            acc = __builtin_amdgcn_wmma_f32_16x16x32_bf16(
                false, afrag[kc], false, bcur[kc], (short)0, acc, false, false);

        // C layout: lane holds col n = lane&15, rows v + 8*(lane>>4).
        // Transpose through LDS so each lane scans one row.
#pragma unroll
        for (int v = 0; v < 8; ++v)
            ctile[wave][v + 8 * half][lrow] = acc[v];
        // wave-synchronous: lanes are lockstep in wave32; wait for ds stores
        asm volatile("s_wait_dscnt 0" ::: "memory");

        // lane owns row lrow, col-half `half`: read its 8 values as 2x b128
        float4 c0 = *(const float4*)&ctile[wave][lrow][half * 8];
        float4 c1 = *(const float4*)&ctile[wave][lrow][half * 8 + 4];
        float cv[8] = {c0.x, c0.y, c0.z, c0.w, c1.x, c1.y, c1.z, c1.w};

        const int i = i0 + lrow;
#pragma unroll
        for (int t = 0; t < 8; ++t) {
            int j = j0 + half * 8 + t;
            bool ok = (j < i) && ((i - j) > 1);
            float val = ok ? cv[t] : NEG_INF;
            if (val > tv[KSEM - 1]) {
                tv[KSEM - 1] = val;
                ti[KSEM - 1] = j;
#pragma unroll
                for (int s = KSEM - 1; s > 0; --s) {
                    if (tv[s] > tv[s - 1]) {
                        float x = tv[s]; tv[s] = tv[s - 1]; tv[s - 1] = x;
                        int   y = ti[s]; ti[s] = ti[s - 1]; ti[s - 1] = y;
                    }
                }
            }
        }

        // rotate double buffer
#pragma unroll
        for (int kc = 0; kc < 8; ++kc) bcur[kc] = bnext[kc];
    }

    // publish per-(wave,half) partial top-5
#pragma unroll
    for (int k = 0; k < KSEM; ++k) {
        svals[lrow][wave * 2 + half][k] = tv[k];
        sidx[lrow][wave * 2 + half][k]  = ti[k];
    }
    __syncthreads();

    // 16 threads merge 8 partials each and emit semantic edges
    if (threadIdx.x < 16) {
        int r = threadIdx.x;
        float fv[KSEM];
        int   fi[KSEM];
#pragma unroll
        for (int k = 0; k < KSEM; ++k) { fv[k] = NEG_INF; fi[k] = -1; }
        for (int p = 0; p < 8; ++p) {
#pragma unroll
            for (int k = 0; k < KSEM; ++k) {
                float val = svals[r][p][k];
                int   id  = sidx[r][p][k];
                if (val > fv[KSEM - 1]) {
                    fv[KSEM - 1] = val;
                    fi[KSEM - 1] = id;
#pragma unroll
                    for (int s = KSEM - 1; s > 0; --s) {
                        if (fv[s] > fv[s - 1]) {
                            float x = fv[s]; fv[s] = fv[s - 1]; fv[s - 1] = x;
                            int   y = fi[s]; fi[s] = fi[s - 1]; fi[s - 1] = y;
                        }
                    }
                }
            }
        }
        const int i = i0 + r;
#pragma unroll
        for (int k = 0; k < KSEM; ++k) {
            int  off   = nT + i * KSEM + k;
            bool valid = fv[k] > THRESH;
            o_ei[off]             = valid ? (float)fi[k] : -1.0f;  // src
            o_ei[(size_t)E + off] = valid ? (float)i     : -1.0f;  // dst
            o_valid[off]          = valid ? 1.0f : 0.0f;
        }
    }
}

// ---------------- Kernel 3: temporal edges ---------------------------------
__global__ __launch_bounds__(256) void mg_temporal(
        float* __restrict__ o_ei, float* __restrict__ o_valid, int N, int E) {
    int e  = blockIdx.x * blockDim.x + threadIdx.x;
    int nT = 2 * (N - 1);
    if (e >= nT) return;
    int p = e >> 1, d = e & 1;
    o_ei[e]             = d ? (float)(p + 1) : (float)p;
    o_ei[(size_t)E + e] = d ? (float)p       : (float)(p + 1);
    o_valid[e]          = 1.0f;
}

// ---------------- Kernel 4: edge_attr gather -------------------------------
__global__ __launch_bounds__(256) void mg_attr(
        const float* __restrict__ table, const float* __restrict__ o_valid,
        float* __restrict__ o_attr, long total, int nT) {
    long idx = (long)blockIdx.x * blockDim.x + threadIdx.x;
    if (idx >= total) return;
    int e = (int)(idx >> 6);
    int c = (int)(idx & 63);
    float v = o_valid[e];
    float t = table[((e >= nT) ? 64 : 0) + c];
    o_attr[idx] = t * v;
}

// ---------------------------------------------------------------------------
extern "C" void kernel_launch(void* const* d_in, const int* in_sizes, int n_in,
                              void* d_out, int out_size, void* d_ws, size_t ws_size,
                              hipStream_t stream) {
    const float* emb   = (const float*)d_in[0];   // [N, 256] f32
    const float* table = (const float*)d_in[1];   // [2, 64]  f32

    int N  = in_sizes[0] / DIM;                   // 8192
    int nT = 2 * (N - 1);                         // 16382
    int E  = nT + N * KSEM;                       // 57342

    unsigned short* ebf = (unsigned short*)d_ws;  // bf16 normalized [N,256]

    float* o_ei    = (float*)d_out;               // [2, E]
    float* o_attr  = o_ei + 2 * (size_t)E;        // [E, 64]
    float* o_valid = o_attr + (size_t)E * 64;     // [E]

    // 1) normalize + bf16 convert (8 rows per 256-thread block)
    mg_normalize_bf16<<<(N + 7) / 8, 256, 0, stream>>>(emb, ebf, N);

    // 2) WMMA similarity + streaming top-5 -> semantic edge_index / valid
    mg_sim_topk<<<N / 16, 128, 0, stream>>>(ebf, o_ei, o_valid, N, nT, E);

    // 3) temporal edge_index / valid
    mg_temporal<<<(nT + 255) / 256, 256, 0, stream>>>(o_ei, o_valid, N, E);

    // 4) edge_attr = table[type] * valid  (reads valid written above)
    long total = (long)E * 64;
    mg_attr<<<(int)((total + 255) / 256), 256, 0, stream>>>(
        table, o_valid, o_attr, total, nT);
}
